// AutoRegressiveGraphGenerator_83339545412144
// MI455X (gfx1250) — matrix operand order repaired and checked by reading, not compiled, not run on hardware
//
#include <hip/hip_runtime.h>
#include <hip/hip_bf16.h>
#include <math.h>

#define N_NODES 256
#define NF      10
#define HD      2048
#define G4H     8192
#define NGG     20
#define KK      10
#define EE      4096

typedef __attribute__((ext_vector_type(16))) __bf16 v16bf;
typedef __attribute__((ext_vector_type(8)))  __bf16 v8bf;
typedef __attribute__((ext_vector_type(8)))  float  v8f;

__device__ __forceinline__ unsigned short f2bf(float f) {
  union { float f; unsigned int u; } x; x.f = f;
  unsigned int u = x.u;
  u += 0x7fffu + ((u >> 16) & 1u);       // round-to-nearest-even
  return (unsigned short)(u >> 16);
}

__device__ __forceinline__ float sigmf(float x) { return 1.0f / (1.0f + __expf(-x)); }

__device__ __forceinline__ v16bf load_afrag(const unsigned short* p) {
  v8bf lo = *(const v8bf*)(const void*)p;
  v8bf hi = *(const v8bf*)(const void*)(p + 8);
  return __builtin_shufflevector(lo, hi, 0,1,2,3,4,5,6,7,8,9,10,11,12,13,14,15);
}
__device__ __forceinline__ v16bf load_bfrag(const unsigned short* p) {
  return *(const v16bf*)(const void*)p;
}

// ---------------- small utility kernels ----------------

__global__ void k_f32_to_bf16(const float* __restrict__ in, unsigned short* __restrict__ out, long n) {
  long i = (long)blockIdx.x * blockDim.x + threadIdx.x;
  if (i < n) out[i] = f2bf(in[i]);
}

__global__ void k_zero32(float* __restrict__ p, int n) {
  int i = blockIdx.x * blockDim.x + threadIdx.x;
  if (i < n) p[i] = 0.0f;
}

__global__ void k_count(const int* __restrict__ dst, float* __restrict__ cnt) {
  int e = blockIdx.x * blockDim.x + threadIdx.x;
  if (e < EE) atomicAdd(&cnt[dst[e]], 1.0f);
}

// scatter-sum x[src] into out[dst] for ngraphs stacked [N, nf] feature maps
__global__ void k_scatter(const float* __restrict__ x, const int* __restrict__ src,
                          const int* __restrict__ dst, float* __restrict__ out,
                          int nf, int ngraphs) {
  long idx = (long)blockIdx.x * blockDim.x + threadIdx.x;
  long total = (long)ngraphs * EE * nf;
  if (idx >= total) return;
  int f = (int)(idx % nf);
  int e = (int)((idx / nf) % EE);
  int k = (int)(idx / ((long)nf * EE));
  atomicAdd(&out[((long)k * N_NODES + dst[e]) * nf + f],
            x[((long)k * N_NODES + src[e]) * nf + f]);
}

// SAGE encoder (dense part) + positional emb, output bf16 [ngraphs, N, HD]
__global__ void k_encode_dense(const float* __restrict__ x, const float* __restrict__ meanbuf,
                               const float* __restrict__ cnt,
                               const float* __restrict__ Wl, const float* __restrict__ bl,
                               const float* __restrict__ Wr, const float* __restrict__ pe,
                               int pebase, int nf, int ngraphs,
                               unsigned short* __restrict__ out) {
  long idx = (long)blockIdx.x * blockDim.x + threadIdx.x;
  long total = (long)ngraphs * N_NODES * HD;
  if (idx >= total) return;
  int h = (int)(idx % HD);
  int n = (int)((idx / HD) % N_NODES);
  int k = (int)(idx / ((long)HD * N_NODES));
  float cn = cnt[n]; cn = cn < 1.0f ? 1.0f : cn;
  float inv = 1.0f / cn;
  float acc = bl[h] + pe[(long)(pebase + k) * HD + h];
  const float* xr = x + ((long)k * N_NODES + n) * nf;
  const float* mr = meanbuf + ((long)k * N_NODES + n) * nf;
  const float* wl = Wl + (long)h * nf;
  const float* wr = Wr + (long)h * nf;
  #pragma unroll
  for (int f = 0; f < NF; ++f)
    acc += (mr[f] * inv) * wl[f] + xr[f] * wr[f];
  out[idx] = f2bf(acc);
}

// ---------------- the LSTM gate GEMM (bf16 WMMA) ----------------
// gates[256, 8192] = X[256,2048] @ Wih^T + H[256,2048] @ Whh^T + b_ih + b_hh
// Block = full M (256 rows) x 64 cols; wave w owns rows 32w..32w+31.
// All 8 waves of a block share the same B fragments each K step -> WGP-cache
// absorbs the 8x reuse; each weight byte crosses L2 once per GEMM.
// Inner loop is double-buffered so next-K fragment loads overlap the 8 WMMAs.
__global__ __launch_bounds__(256) void k_gates_wmma(
    const unsigned short* __restrict__ X,
    const unsigned short* __restrict__ Hb,
    const unsigned short* __restrict__ Wih,
    const unsigned short* __restrict__ Whh,
    const float* __restrict__ bih,
    const float* __restrict__ bhh,
    float* __restrict__ g) {
  const int lane = threadIdx.x & 31;
  const int wave = threadIdx.x >> 5;
  const int hl = lane & 15;
  const int hi = lane >> 4;

  const int m0 = wave * 32;               // 32-row sub-tile per wave
  const int n0 = blockIdx.x * 64;         // 64-col tile per block

  const int akoff = hi * 16;              // A frag K base: contiguous k = 16*hi + s
  const int bkoff = hi * 16;              // B frag K base: contiguous k = 16*hi + s

  v8f acc[8] = {};                        // acc[2*j + r]: rows m0+16r, cols n0+16j

  #pragma unroll
  for (int pass = 0; pass < 2; ++pass) {
    const unsigned short* A = pass ? Hb : X;
    const unsigned short* W = pass ? Whh : Wih;
    const unsigned short* ap0 = A + (size_t)(m0 + hl) * HD + akoff;
    const unsigned short* ap1 = A + (size_t)(m0 + 16 + hl) * HD + akoff;
    const unsigned short* bp0 = W + (size_t)(n0 +  0 + hl) * HD + bkoff;
    const unsigned short* bp1 = W + (size_t)(n0 + 16 + hl) * HD + bkoff;
    const unsigned short* bp2 = W + (size_t)(n0 + 32 + hl) * HD + bkoff;
    const unsigned short* bp3 = W + (size_t)(n0 + 48 + hl) * HD + bkoff;

    // prologue: current fragments for k0 = 0
    v16bf a0c = load_afrag(ap0);
    v16bf a1c = load_afrag(ap1);
    v16bf b0c = load_bfrag(bp0);
    v16bf b1c = load_bfrag(bp1);
    v16bf b2c = load_bfrag(bp2);
    v16bf b3c = load_bfrag(bp3);

    for (int k0 = 0; k0 < HD; k0 += 32) {
      // prefetch next K-step (dummy re-load of k=32 on last iter; discarded)
      const int kn = (k0 + 32 < HD) ? (k0 + 32) : 32;
      v16bf a0n = load_afrag(ap0 + kn);
      v16bf a1n = load_afrag(ap1 + kn);
      v16bf b0n = load_bfrag(bp0 + kn);
      v16bf b1n = load_bfrag(bp1 + kn);
      v16bf b2n = load_bfrag(bp2 + kn);
      v16bf b3n = load_bfrag(bp3 + kn);

      acc[0] = __builtin_amdgcn_wmma_f32_16x16x32_bf16(false, a0c, false, b0c, (short)0, acc[0], false, false);
      acc[1] = __builtin_amdgcn_wmma_f32_16x16x32_bf16(false, a1c, false, b0c, (short)0, acc[1], false, false);
      acc[2] = __builtin_amdgcn_wmma_f32_16x16x32_bf16(false, a0c, false, b1c, (short)0, acc[2], false, false);
      acc[3] = __builtin_amdgcn_wmma_f32_16x16x32_bf16(false, a1c, false, b1c, (short)0, acc[3], false, false);
      acc[4] = __builtin_amdgcn_wmma_f32_16x16x32_bf16(false, a0c, false, b2c, (short)0, acc[4], false, false);
      acc[5] = __builtin_amdgcn_wmma_f32_16x16x32_bf16(false, a1c, false, b2c, (short)0, acc[5], false, false);
      acc[6] = __builtin_amdgcn_wmma_f32_16x16x32_bf16(false, a0c, false, b3c, (short)0, acc[6], false, false);
      acc[7] = __builtin_amdgcn_wmma_f32_16x16x32_bf16(false, a1c, false, b3c, (short)0, acc[7], false, false);

      a0c = a0n; a1c = a1n;
      b0c = b0n; b1c = b1n; b2c = b2n; b3c = b3n;
    }
  }

  #pragma unroll
  for (int r = 0; r < 2; ++r) {
    const int crow = m0 + 16 * r + hi * 8;    // C/D layout: vgpr v -> row crow+v
    #pragma unroll
    for (int j = 0; j < 4; ++j) {
      const int col = n0 + 16 * j + hl;
      const float bias = bih[col] + bhh[col];
      float* gp = g + (size_t)crow * G4H + col;
      #pragma unroll
      for (int v = 0; v < 8; ++v)
        gp[(size_t)v * G4H] = acc[2 * j + r][v] + bias;
    }
  }
}

// pointwise LSTM recurrence; gate order i,f,g,o; emits h in f32 and bf16
__global__ void k_lstm(const float* __restrict__ g, float* __restrict__ h,
                       float* __restrict__ c, unsigned short* __restrict__ hbf) {
  int idx = blockIdx.x * blockDim.x + threadIdx.x;
  if (idx >= N_NODES * HD) return;
  int n = idx >> 11;
  int d = idx & (HD - 1);
  const float* gr = g + (size_t)n * G4H;
  float gi = sigmf(gr[d]);
  float gf = sigmf(gr[HD + d]);
  float gg = tanhf(gr[2 * HD + d]);
  float go = sigmf(gr[3 * HD + d]);
  float c2 = gf * c[idx] + gi * gg;
  float h2 = go * tanhf(c2);
  c[idx] = c2;
  h[idx] = h2;
  hbf[idx] = f2bf(h2);
}

// decoder projections: y = h + pe[row];  pl = y @ Wdl^T, pr = y @ Wdr^T  (out dim 8)
__global__ void k_decode_p(const float* __restrict__ h, const float* __restrict__ pe, int perow,
                           const float* __restrict__ Wdl, const float* __restrict__ Wdr,
                           float* __restrict__ pl, float* __restrict__ pr) {
  int idx = blockIdx.x * blockDim.x + threadIdx.x;
  if (idx >= N_NODES * 8) return;
  int o = idx & 7;
  int n = idx >> 3;
  const float* hr  = h  + (size_t)n * HD;
  const float* per = pe + (size_t)perow * HD;
  const float* wl  = Wdl + (size_t)o * HD;
  const float* wr  = Wdr + (size_t)o * HD;
  float al = 0.0f, ar = 0.0f;
  for (int k = 0; k < HD; ++k) {
    float y = hr[k] + per[k];
    al += y * wl[k];
    ar += y * wr[k];
  }
  pl[idx] = al;
  pr[idx] = ar;
}

__global__ void k_scatter8(const float* __restrict__ pl, const int* __restrict__ src,
                           const int* __restrict__ dst, float* __restrict__ agg) {
  int idx = blockIdx.x * blockDim.x + threadIdx.x;
  if (idx >= EE * 8) return;
  int o = idx & 7;
  int e = idx >> 3;
  atomicAdd(&agg[dst[e] * 8 + o], pl[src[e] * 8 + o]);
}

// x_next = [static2 | mean(pl)/cnt + b_dec_l + pr]; also writes the output slice
__global__ void k_xnext(const float* __restrict__ agg, const float* __restrict__ cnt,
                        const float* __restrict__ bdl, const float* __restrict__ pr,
                        const float* __restrict__ lastknown, float* __restrict__ xnext,
                        float* __restrict__ outslice) {
  int idx = blockIdx.x * blockDim.x + threadIdx.x;
  if (idx >= N_NODES * NF) return;
  int f = idx % NF;
  int n = idx / NF;
  float v;
  if (f < 2) {
    v = lastknown[n * NF + f];
  } else {
    int o = f - 2;
    float cn = cnt[n]; cn = cn < 1.0f ? 1.0f : cn;
    v = agg[n * 8 + o] / cn + bdl[o] + pr[n * 8 + o];
  }
  xnext[idx] = v;
  outslice[idx] = v;
}

// ---------------- host orchestration ----------------

static inline int nblk(long n, int b) { return (int)((n + b - 1) / b); }

extern "C" void kernel_launch(void* const* d_in, const int* in_sizes, int n_in,
                              void* d_out, int out_size, void* d_ws, size_t ws_size,
                              hipStream_t stream) {
  (void)in_sizes; (void)n_in; (void)out_size; (void)ws_size;
  const float* known = (const float*)d_in[0];
  const int*   ei    = (const int*)d_in[1];
  const float* Wencl = (const float*)d_in[2];
  const float* bencl = (const float*)d_in[3];
  const float* Wencr = (const float*)d_in[4];
  const float* pe    = (const float*)d_in[5];
  const float* Wih   = (const float*)d_in[6];
  const float* Whh   = (const float*)d_in[7];
  const float* bih   = (const float*)d_in[8];
  const float* bhh   = (const float*)d_in[9];
  const float* Wdl   = (const float*)d_in[10];
  const float* bdl   = (const float*)d_in[11];
  const float* Wdr   = (const float*)d_in[12];
  float* out = (float*)d_out;
  const int* src = ei;
  const int* dst = ei + EE;

  char* ws = (char*)d_ws;
  size_t off = 0;
  auto take = [&](size_t bytes) -> char* {
    char* p = ws + off;
    off = (off + bytes + 255) & ~(size_t)255;
    return p;
  };

  unsigned short* Wih_bf = (unsigned short*)take((size_t)G4H * HD * 2);
  unsigned short* Whh_bf = (unsigned short*)take((size_t)G4H * HD * 2);
  unsigned short* hidbf  = (unsigned short*)take((size_t)KK * N_NODES * HD * 2);
  float* hbuf  = (float*)take((size_t)N_NODES * HD * 4);
  float* cbuf  = (float*)take((size_t)N_NODES * HD * 4);
  unsigned short* hbf = (unsigned short*)take((size_t)N_NODES * HD * 2);
  unsigned short* xbf = (unsigned short*)take((size_t)N_NODES * HD * 2);
  float* gates = (float*)take((size_t)N_NODES * G4H * 4);
  float* cnt   = (float*)take((size_t)N_NODES * 4);
  float* meanK = (float*)take((size_t)KK * N_NODES * NF * 4);
  float* meanx = (float*)take((size_t)N_NODES * NF * 4);
  float* pl    = (float*)take((size_t)N_NODES * 8 * 4);
  float* pr    = (float*)take((size_t)N_NODES * 8 * 4);
  float* agg   = (float*)take((size_t)N_NODES * 8 * 4);
  float* xnext = (float*)take((size_t)N_NODES * NF * 4);

  const int TB = 256;
  const long nW = (long)G4H * HD;

  // one-time prep
  k_f32_to_bf16<<<nblk(nW, TB), TB, 0, stream>>>(Wih, Wih_bf, nW);
  k_f32_to_bf16<<<nblk(nW, TB), TB, 0, stream>>>(Whh, Whh_bf, nW);
  k_zero32<<<nblk(N_NODES, TB), TB, 0, stream>>>(cnt, N_NODES);
  k_count<<<nblk(EE, TB), TB, 0, stream>>>(dst, cnt);
  k_zero32<<<nblk((long)KK * N_NODES * NF, TB), TB, 0, stream>>>(meanK, KK * N_NODES * NF);
  k_scatter<<<nblk((long)KK * EE * NF, TB), TB, 0, stream>>>(known, src, dst, meanK, NF, KK);
  k_encode_dense<<<nblk((long)KK * N_NODES * HD, TB), TB, 0, stream>>>(
      known, meanK, cnt, Wencl, bencl, Wencr, pe, 0, NF, KK, hidbf);
  k_zero32<<<nblk((long)N_NODES * HD, TB), TB, 0, stream>>>(hbuf, N_NODES * HD);
  k_zero32<<<nblk((long)N_NODES * HD, TB), TB, 0, stream>>>(cbuf, N_NODES * HD);
  k_zero32<<<nblk((long)N_NODES * HD / 2, TB), TB, 0, stream>>>((float*)hbf, N_NODES * HD / 2);

  // LSTM warm-up over K known layers
  for (int t = 0; t < KK; ++t) {
    k_gates_wmma<<<128, 256, 0, stream>>>(hidbf + (size_t)t * N_NODES * HD, hbf,
                                          Wih_bf, Whh_bf, bih, bhh, gates);
    k_lstm<<<nblk((long)N_NODES * HD, TB), TB, 0, stream>>>(gates, hbuf, cbuf, hbf);
  }

  // autoregressive generation
  for (int t = 0; t < NGG - KK; ++t) {
    k_decode_p<<<nblk((long)N_NODES * 8, TB), TB, 0, stream>>>(hbuf, pe, KK + t, Wdl, Wdr, pl, pr);
    k_zero32<<<nblk((long)N_NODES * 8, TB), TB, 0, stream>>>(agg, N_NODES * 8);
    k_scatter8<<<nblk((long)EE * 8, TB), TB, 0, stream>>>(pl, src, dst, agg);
    k_xnext<<<nblk((long)N_NODES * NF, TB), TB, 0, stream>>>(
        agg, cnt, bdl, pr, known + (size_t)(KK - 1) * N_NODES * NF, xnext,
        out + (size_t)t * N_NODES * NF);
    k_zero32<<<nblk((long)N_NODES * NF, TB), TB, 0, stream>>>(meanx, N_NODES * NF);
    k_scatter<<<nblk((long)EE * NF, TB), TB, 0, stream>>>(xnext, src, dst, meanx, NF, 1);
    k_encode_dense<<<nblk((long)N_NODES * HD, TB), TB, 0, stream>>>(
        xnext, meanx, cnt, Wencl, bencl, Wencr, pe, KK + t, NF, 1, xbf);
    k_gates_wmma<<<128, 256, 0, stream>>>(xbf, hbf, Wih_bf, Whh_bf, bih, bhh, gates);
    k_lstm<<<nblk((long)N_NODES * HD, TB), TB, 0, stream>>>(gates, hbuf, cbuf, hbf);
  }
}
